// Encoder_37177236914872
// MI455X (gfx1250) — matrix operand order repaired and checked by reading, not compiled, not run on hardware
//
#include <hip/hip_runtime.h>
#include <hip/hip_bf16.h>

typedef __bf16 bf16;
typedef __attribute__((ext_vector_type(16))) __bf16 v16bf;
typedef __attribute__((ext_vector_type(8)))  __bf16 v8bf;
typedef __attribute__((ext_vector_type(8)))  float  v8f;

#define HB 256                 // hidden width
#define NBN 128                // nodes
#define BBB 8                  // batch
#define EE 16256               // edges = N*(N-1)
#define MEDGE (BBB*EE)         // 130048 edge rows
#define MNODE (BBB*NBN)        // 1024 node rows

__device__ __forceinline__ float elu_f(float x) {
  return x > 0.f ? x : (__expf(x) - 1.f);
}

// A-fragment for v_wmma_f32_16x16x32_bf16 from a row-major bf16 row pointer.
// Layout (16-bit A 16x32): lanes 0-15: elems0-7=K0..7, elems8-15=K16..23
//                          lanes16-31: elems0-7=K8..15, elems8-15=K24..31
__device__ __forceinline__ v16bf afrag_bf(const bf16* rb, int half) {
  v8bf c0 = *(const v8bf*)(rb + half * 8);
  v8bf c1 = *(const v8bf*)(rb + 16 + half * 8);
  v16bf a;
#pragma unroll
  for (int i = 0; i < 8; ++i) { a[i] = c0[i]; a[i + 8] = c1[i]; }
  return a;
}

__device__ __forceinline__ v16bf afrag_f32(const float* rb, int half) {
  v8f c0 = *(const v8f*)(rb + half * 8);
  v8f c1 = *(const v8f*)(rb + 16 + half * 8);
  v16bf a;
#pragma unroll
  for (int i = 0; i < 8; ++i) { a[i] = (bf16)c0[i]; a[i + 8] = (bf16)c1[i]; }
  return a;
}

// Stage a 32(K) x 256(col) weight chunk (16KB) straight into LDS with the
// gfx1250 async-copy path (ASYNCcnt-tracked, no VGPR staging).
// 1024 x 16B transfers; 256 threads x 4 async instructions each.
__device__ __forceinline__ void stage_wchunk(const bf16* __restrict__ Wt, int Kpad,
                                             int k0, bf16 (*wbuf)[32], int tid) {
#pragma unroll
  for (int j = 0; j < 4; ++j) {
    const int id  = j * 256 + tid;   // 0..1023
    const int col = id >> 2;
    const int sub = id & 3;
    const unsigned long long g =
        (unsigned long long)(uintptr_t)(Wt + (size_t)col * Kpad + k0 + sub * 8);
    const unsigned lds = (unsigned)(uintptr_t)(&wbuf[col][sub * 8]);
    asm volatile("global_load_async_to_lds_b128 %0, %1, off"
                 :: "v"(lds), "v"(g) : "memory");
  }
}

// Fused 2-layer MLP (both layers output HB=256 cols), optional gather-A and
// optional final 256->2 projection.
// Block: 256 threads = 8 waves. Waves 0-3 own cols 0-127 of row-groups 0-3,
// waves 4-7 own cols 128-255. Each wave: 32 rows (2 subtiles) x 128 cols
// -> 16 fp32 accum tiles; every LDS B-fragment feeds 2 WMMAs.
// MODE 0: A = bf16 [M, strideA]            (node MLP1 on padded x)
// MODE 1: A = fp32 [M, strideA]            (node MLP2 on aggregated sums)
// MODE 2: A = [h[recv], h[send]]  (K=512)  (edge MLP1) -> outb
// MODE 3: A = [h2[recv], h2[send], skip]   (edge MLP2) -> proj -> outf
template<int MODE, int K1STEPS>
__global__ __launch_bounds__(256) void fused_mlp(
    const bf16* __restrict__ Abf, const float* __restrict__ Af, int strideA,
    const int* __restrict__ recv, const int* __restrict__ send,
    const bf16* __restrict__ skip,
    const bf16* __restrict__ W1t, const float* __restrict__ B1,   // W1t: [256][K1pad] col-major bf16
    const bf16* __restrict__ W2t, const float* __restrict__ B2,   // W2t: [256][256]
    bf16* __restrict__ outb,
    const float* __restrict__ OW, const float* __restrict__ OB,
    float* __restrict__ outf)
{
  __shared__ __align__(64) bf16 wtile[2][256][32];   // double-buffered weight chunk, 32KB
  __shared__ __align__(64) bf16 itile[4][32][256];   // layer-1 out per row-group, 64KB

  const int tid  = threadIdx.x;
  const int wave = tid >> 5;
  const int lane = tid & 31;
  const int lr   = lane & 15;     // row-in-tile (A) / col-in-tile (B,C)
  const int half = lane >> 4;
  const int grp  = wave & 3;      // row-group 0..3
  const int nc0  = (wave >> 2) * 128;            // this wave's first column

  const int brow0 = blockIdx.x * 128;            // first row of block
  const int row0  = brow0 + grp * 32;            // first row of this wave's group

  // per-lane A segment bases for the 2 subtiles (each segment row = HB elems)
  const bf16*  rbase[2] = {nullptr, nullptr};
  const bf16*  sbase[2] = {nullptr, nullptr};
  const bf16*  kbase[2] = {nullptr, nullptr};
  const bf16*  nbase[2] = {nullptr, nullptr};
  const float* fbase[2] = {nullptr, nullptr};
#pragma unroll
  for (int st = 0; st < 2; ++st) {
    const int mr = row0 + st * 16 + lr;
    if constexpr (MODE == 0) nbase[st] = Abf + (size_t)mr * strideA;
    if constexpr (MODE == 1) fbase[st] = Af + (size_t)mr * strideA;
    if constexpr (MODE >= 2) {
      const int b = mr / EE;
      const int e = mr - b * EE;
      rbase[st] = Abf + ((size_t)(b * NBN + recv[e])) * HB;
      sbase[st] = Abf + ((size_t)(b * NBN + send[e])) * HB;
      if constexpr (MODE == 3) kbase[st] = skip + (size_t)mr * HB;
    }
  }

  v8f acc[2][8];
#pragma unroll
  for (int st = 0; st < 2; ++st)
#pragma unroll
    for (int nt = 0; nt < 8; ++nt)
#pragma unroll
      for (int i = 0; i < 8; ++i) acc[st][nt][i] = 0.f;

  // ---------------- layer 1: K = K1STEPS*32 ----------------
  const int K1pad = K1STEPS * 32;
  stage_wchunk(W1t, K1pad, 0, wtile[0], tid);
  for (int ks = 0; ks < K1STEPS; ++ks) {
    const int bb = ks & 1;
    const int k0 = ks * 32;
    if (ks + 1 < K1STEPS) {
      stage_wchunk(W1t, K1pad, k0 + 32, wtile[bb ^ 1], tid);
      asm volatile("s_wait_asynccnt 0x4" ::: "memory");  // retire current buffer
    } else {
      asm volatile("s_wait_asynccnt 0x0" ::: "memory");
    }
    __syncthreads();

    v16bf a0, a1;
    if constexpr (MODE == 0) {
      a0 = afrag_bf(nbase[0] + k0, half);
      a1 = afrag_bf(nbase[1] + k0, half);
    } else if constexpr (MODE == 1) {
      a0 = afrag_f32(fbase[0] + k0, half);
      a1 = afrag_f32(fbase[1] + k0, half);
    } else if constexpr (MODE == 2) {
      a0 = afrag_bf(k0 < HB ? rbase[0] + k0 : sbase[0] + (k0 - HB), half);
      a1 = afrag_bf(k0 < HB ? rbase[1] + k0 : sbase[1] + (k0 - HB), half);
    } else {
      const bf16* b0 = (k0 < HB) ? (rbase[0] + k0)
                     : (k0 < 2 * HB ? sbase[0] + (k0 - HB) : kbase[0] + (k0 - 2 * HB));
      const bf16* b1 = (k0 < HB) ? (rbase[1] + k0)
                     : (k0 < 2 * HB ? sbase[1] + (k0 - HB) : kbase[1] + (k0 - 2 * HB));
      a0 = afrag_bf(b0, half);
      a1 = afrag_bf(b1, half);
    }
#pragma unroll
    for (int nt = 0; nt < 8; ++nt) {
      v16bf bfr = afrag_bf(&wtile[bb][nc0 + nt * 16 + lr][0], half);
      acc[0][nt] = __builtin_amdgcn_wmma_f32_16x16x32_bf16(
          false, a0, false, bfr, (short)0, acc[0][nt], false, false);
      acc[1][nt] = __builtin_amdgcn_wmma_f32_16x16x32_bf16(
          false, a1, false, bfr, (short)0, acc[1][nt], false, false);
    }
    __syncthreads();   // all waves done reading wtile[bb] before it is restaged
  }

  // layer-1 epilogue: bias + ELU -> row-group LDS tile (bf16)
  bf16 (*my)[256] = itile[grp];
#pragma unroll
  for (int st = 0; st < 2; ++st)
#pragma unroll
    for (int nt = 0; nt < 8; ++nt) {
      const int col = nc0 + nt * 16 + lr;
      const float bc = B1[col];
#pragma unroll
      for (int v = 0; v < 8; ++v)
        my[st * 16 + half * 8 + v][col] = (bf16)elu_f(acc[st][nt][v] + bc);
    }

  // ---------------- layer 2: K = 256 ----------------
#pragma unroll
  for (int st = 0; st < 2; ++st)
#pragma unroll
    for (int nt = 0; nt < 8; ++nt)
#pragma unroll
      for (int i = 0; i < 8; ++i) acc[st][nt][i] = 0.f;

  stage_wchunk(W2t, HB, 0, wtile[0], tid);
  for (int ks = 0; ks < 8; ++ks) {
    const int bb = ks & 1;
    const int k0 = ks * 32;
    if (ks + 1 < 8) {
      stage_wchunk(W2t, HB, k0 + 32, wtile[bb ^ 1], tid);
      asm volatile("s_wait_asynccnt 0x4" ::: "memory");
    } else {
      asm volatile("s_wait_asynccnt 0x0" ::: "memory");
    }
    __syncthreads();   // also makes layer-1 epilogue visible across the wave pair

    v16bf a0 = afrag_bf(&my[lr][k0], half);
    v16bf a1 = afrag_bf(&my[16 + lr][k0], half);
#pragma unroll
    for (int nt = 0; nt < 8; ++nt) {
      v16bf bfr = afrag_bf(&wtile[bb][nc0 + nt * 16 + lr][0], half);
      acc[0][nt] = __builtin_amdgcn_wmma_f32_16x16x32_bf16(
          false, a0, false, bfr, (short)0, acc[0][nt], false, false);
      acc[1][nt] = __builtin_amdgcn_wmma_f32_16x16x32_bf16(
          false, a1, false, bfr, (short)0, acc[1][nt], false, false);
    }
    __syncthreads();
  }

  if constexpr (MODE != 3) {
#pragma unroll
    for (int st = 0; st < 2; ++st)
#pragma unroll
      for (int nt = 0; nt < 8; ++nt) {
        const int col = nc0 + nt * 16 + lr;
        const float bc = B2[col];
#pragma unroll
        for (int v = 0; v < 8; ++v) {
          const int rowg = row0 + st * 16 + half * 8 + v;
          outb[(size_t)rowg * HB + col] = (bf16)elu_f(acc[st][nt][v] + bc);
        }
      }
  } else {
    // layer-2 epilogue back into the row-group LDS tile
#pragma unroll
    for (int st = 0; st < 2; ++st)
#pragma unroll
      for (int nt = 0; nt < 8; ++nt) {
        const int col = nc0 + nt * 16 + lr;
        const float bc = B2[col];
#pragma unroll
        for (int v = 0; v < 8; ++v)
          my[st * 16 + half * 8 + v][col] = (bf16)elu_f(acc[st][nt][v] + bc);
      }
    __syncthreads();   // full 256-wide rows are written by two waves
    // final projection 256 -> 2: wave (grp, h=wave>>2) covers col h of its
    // 32 rows; one dot product per lane.
    const int pcol = wave >> 2;
    float po = 0.f;
#pragma unroll 8
    for (int f = 0; f < HB; ++f)
      po += (float)my[lane][f] * OW[f * 2 + pcol];
    outf[(size_t)(row0 + lane) * 2 + pcol] = po + OB[pcol];
  }
}

// ---------------- auxiliary kernels ----------------

// derive integer receiver/sender indices from one-hot [E, N] matrices
__global__ void idx_kernel(const float* __restrict__ rec, const float* __restrict__ snd,
                           int* __restrict__ recv, int* __restrict__ send) {
  const int e = blockIdx.x * blockDim.x + threadIdx.x;
  if (e >= EE) return;
  int r = 0, s = 0;
  for (int n = 0; n < NBN; ++n) {
    if (rec[(size_t)e * NBN + n] > 0.5f) r = n;
    if (snd[(size_t)e * NBN + n] > 0.5f) s = n;
  }
  recv[e] = r; send[e] = s;
}

// transpose + convert + K-pad weights: fp32 [K, 256] -> bf16 [256, Kpad]
__global__ void wprep_kernel(const float* __restrict__ w, bf16* __restrict__ wt,
                             int K, int Kpad) {
  const int id = blockIdx.x * blockDim.x + threadIdx.x;
  if (id >= 256 * Kpad) return;
  const int col = id / Kpad, k = id - col * Kpad;
  wt[id] = (k < K) ? (bf16)w[(size_t)k * 256 + col] : (bf16)0.f;
}

// convert + K-pad activations: fp32 [M, K] -> bf16 [M, Kpad]
__global__ void xprep_kernel(const float* __restrict__ x, bf16* __restrict__ xb,
                             int M, int K, int Kpad) {
  const int id = blockIdx.x * blockDim.x + threadIdx.x;
  if (id >= M * Kpad) return;
  const int row = id / Kpad, k = id - row * Kpad;
  xb[id] = (k < K) ? (bf16)x[(size_t)row * K + k] : (bf16)0.f;
}

__global__ void zero_kernel(float* __restrict__ p, int n) {
  const int id = blockIdx.x * blockDim.x + threadIdx.x;
  if (id < n) p[id] = 0.f;
}

// scatter-add edge features into receiver nodes: agg[b, recv[e], :] += e1[b, e, :]
__global__ void agg_kernel(const bf16* __restrict__ e1, const int* __restrict__ recv,
                           float* __restrict__ agg) {
  const int row = blockIdx.x;        // global edge row in [0, B*E)
  const int col = threadIdx.x;       // 0..255
  const int b = row / EE;
  const int e = row - b * EE;
  atomicAdd(&agg[((size_t)(b * NBN + recv[e])) * HB + col],
            (float)e1[(size_t)row * HB + col]);
}

extern "C" void kernel_launch(void* const* d_in, const int* in_sizes, int n_in,
                              void* d_out, int out_size, void* d_ws, size_t ws_size,
                              hipStream_t stream) {
  (void)in_sizes; (void)n_in; (void)out_size; (void)ws_size;
  const float* x        = (const float*)d_in[0];
  const float* rec_rel  = (const float*)d_in[1];
  const float* send_rel = (const float*)d_in[2];
  const float* n1w1 = (const float*)d_in[3];  const float* n1b1 = (const float*)d_in[4];
  const float* n1w2 = (const float*)d_in[5];  const float* n1b2 = (const float*)d_in[6];
  const float* e1w1 = (const float*)d_in[7];  const float* e1b1 = (const float*)d_in[8];
  const float* e1w2 = (const float*)d_in[9];  const float* e1b2 = (const float*)d_in[10];
  const float* n2w1 = (const float*)d_in[11]; const float* n2b1 = (const float*)d_in[12];
  const float* n2w2 = (const float*)d_in[13]; const float* n2b2 = (const float*)d_in[14];
  const float* e2w1 = (const float*)d_in[15]; const float* e2b1 = (const float*)d_in[16];
  const float* e2w2 = (const float*)d_in[17]; const float* e2b2 = (const float*)d_in[18];
  const float* ow   = (const float*)d_in[19]; const float* ob   = (const float*)d_in[20];
  float* out = (float*)d_out;

  char* ws = (char*)d_ws;
  size_t off = 0;
  auto carve = [&](size_t bytes) -> void* {
    void* p = ws + off;
    off += (bytes + 255) & ~(size_t)255;
    return p;
  };

  bf16* xb   = (bf16*)carve((size_t)MNODE * 224 * 2);
  bf16* wn1a = (bf16*)carve(256 * 224 * 2);
  bf16* wn1b = (bf16*)carve(256 * 256 * 2);
  bf16* we1a = (bf16*)carve(256 * 512 * 2);
  bf16* we1b = (bf16*)carve(256 * 256 * 2);
  bf16* wn2a = (bf16*)carve(256 * 256 * 2);
  bf16* wn2b = (bf16*)carve(256 * 256 * 2);
  bf16* we2a = (bf16*)carve(256 * 768 * 2);
  bf16* we2b = (bf16*)carve(256 * 256 * 2);
  int*  recv = (int*)carve(EE * 4);
  int*  send = (int*)carve(EE * 4);
  bf16* h1   = (bf16*)carve((size_t)MNODE * HB * 2);
  bf16* h2   = (bf16*)carve((size_t)MNODE * HB * 2);
  float* agg = (float*)carve((size_t)MNODE * HB * 4);
  bf16* e1   = (bf16*)carve((size_t)MEDGE * HB * 2);

  // ---- prep ----
  idx_kernel<<<(EE + 255) / 256, 256, 0, stream>>>(rec_rel, send_rel, recv, send);
  xprep_kernel<<<(MNODE * 224 + 255) / 256, 256, 0, stream>>>(x, xb, MNODE, 196, 224);
  wprep_kernel<<<(256 * 224 + 255) / 256, 256, 0, stream>>>(n1w1, wn1a, 196, 224);
  wprep_kernel<<<(256 * 256 + 255) / 256, 256, 0, stream>>>(n1w2, wn1b, 256, 256);
  wprep_kernel<<<(256 * 512 + 255) / 256, 256, 0, stream>>>(e1w1, we1a, 512, 512);
  wprep_kernel<<<(256 * 256 + 255) / 256, 256, 0, stream>>>(e1w2, we1b, 256, 256);
  wprep_kernel<<<(256 * 256 + 255) / 256, 256, 0, stream>>>(n2w1, wn2a, 256, 256);
  wprep_kernel<<<(256 * 256 + 255) / 256, 256, 0, stream>>>(n2w2, wn2b, 256, 256);
  wprep_kernel<<<(256 * 768 + 255) / 256, 256, 0, stream>>>(e2w1, we2a, 768, 768);
  wprep_kernel<<<(256 * 256 + 255) / 256, 256, 0, stream>>>(e2w2, we2b, 256, 256);

  // ---- node MLP1: [1024, 224] -> h1 [1024, 256] ----
  fused_mlp<0, 7><<<MNODE / 128, 256, 0, stream>>>(
      xb, nullptr, 224, nullptr, nullptr, nullptr,
      wn1a, n1b1, wn1b, n1b2, h1, nullptr, nullptr, nullptr);

  // ---- edge MLP1 (gather-fused): [130048, 512] -> e1 [130048, 256] ----
  fused_mlp<2, 16><<<MEDGE / 128, 256, 0, stream>>>(
      h1, nullptr, 0, recv, send, nullptr,
      we1a, e1b1, we1b, e1b2, e1, nullptr, nullptr, nullptr);

  // ---- aggregate: agg[b, n, :] = sum_{recv[e]==n} e1[b, e, :] ----
  zero_kernel<<<(MNODE * HB + 255) / 256, 256, 0, stream>>>(agg, MNODE * HB);
  agg_kernel<<<MEDGE, 256, 0, stream>>>(e1, recv, agg);

  // ---- node MLP2: agg fp32 [1024, 256] -> h2 [1024, 256] ----
  fused_mlp<1, 8><<<MNODE / 128, 256, 0, stream>>>(
      nullptr, agg, 256, nullptr, nullptr, nullptr,
      wn2a, n2b1, wn2b, n2b2, h2, nullptr, nullptr, nullptr);

  // ---- edge MLP2 (gather+skip fused) + output projection -> out [130048, 2] ----
  fused_mlp<3, 24><<<MEDGE / 128, 256, 0, stream>>>(
      h2, nullptr, 0, recv, send, e1,
      we2a, e2b1, we2b, e2b2, nullptr, ow, ob, out);
}
